// Qwen2DyMemDecoderLayer_14559939133668
// MI455X (gfx1250) — matrix operand order repaired and checked by reading, not compiled, not run on hardware
//
#include <hip/hip_runtime.h>
#include <hip/hip_bf16.h>
#include <cstdint>

// ---------------- problem constants ----------------
constexpr int Bq   = 1;
constexpr int T    = 2048;
constexpr int H    = 1024;
constexpr int NH   = 16;
constexpr int HD   = 64;
constexpr int NKV  = 4;
constexpr int G    = NH / NKV;           // 4
constexpr int SINK = 128;
constexpr int WIN  = 256;
constexpr int STATE= 256;
constexpr int KSIZE= 4;
constexpr int MINT = H;                  // 1024
constexpr int CONV_DIM = MINT + 2 * G * STATE;      // 3072
constexpr int D_INPROJ = MINT + CONV_DIM + NH;      // 4112
constexpr int INTER = 4096;
constexpr int Lm   = T - WIN - SINK;     // 1664
constexpr int Tc   = SINK + 1 + WIN;     // 385
constexpr float EPS = 1e-6f;

typedef __bf16 bf16;
typedef __attribute__((ext_vector_type(16))) __bf16 v16bf;
typedef __attribute__((ext_vector_type(8)))  float  v8f;

union ABfrag { uint4 u[2]; v16bf v; };

// ---------------- GEMM: C[M,N] = A[M,K](bf16,row) @ Bt[N,K](bf16) + bias ----
// one 64x64 C tile per wave: 4x4 grid of v_wmma_f32_16x16x32_bf16 accumulators.
// Per k-step(32): 8 fragment loads (4 A + 4 B, two b128 each) feed 16 WMMAs
// -> 1 b128 load per WMMA (4x the flops/byte of a single-tile wave).
__global__ __launch_bounds__(256)
void gemm_bf16_wmma(const bf16* __restrict__ A, const bf16* __restrict__ Bt,
                    const float* __restrict__ bias, float* __restrict__ C,
                    int M, int N, int K, int tilesM, int tilesN) {
  const int wave = threadIdx.x >> 5;
  const int lane = threadIdx.x & 31;
  const int tile = blockIdx.x * (blockDim.x >> 5) + wave;
  if (tile >= tilesM * tilesN) return;            // wave-uniform exit
  const int tm = tile / tilesN, tn = tile % tilesN;
  const int hi = lane >> 4;                       // half-wave select
  const int lr = lane & 15;

  const bf16* arow[4];
  const bf16* bcol[4];
#pragma unroll
  for (int i = 0; i < 4; ++i) {
    int mA = tm * 64 + i * 16 + lr; if (mA >= M) mA = M - 1;  // clamp loads
    int nB = tn * 64 + i * 16 + lr; if (nB >= N) nB = N - 1;
    arow[i] = A  + (size_t)mA * K;
    bcol[i] = Bt + (size_t)nB * K;
  }

  v8f acc[4][4] = {};
  for (int k0 = 0; k0 < K; k0 += 32) {
    ABfrag af[4], bfr[4];
#pragma unroll
    for (int i = 0; i < 4; ++i) {
      __builtin_prefetch(arow[i] + k0 + 256, 0, 1);
      __builtin_prefetch(bcol[i] + k0 + 256, 0, 1);
      // A 16x32 frag: lanes 0-15 K k0+{0..7,16..23}; lanes 16-31 K k0+{8..15,24..31}
      const uint4* pa = reinterpret_cast<const uint4*>(arow[i] + k0 + hi * 8);
      af[i].u[0] = pa[0]; af[i].u[1] = pa[2];
      // B 32x16 frag: lanes 0-15 K k0+{0..7,16..23}; lanes 16-31 K k0+{8..15,24..31}
      const uint4* pb = reinterpret_cast<const uint4*>(bcol[i] + k0 + hi * 8);
      bfr[i].u[0] = pb[0]; bfr[i].u[1] = pb[2];
    }
#pragma unroll
    for (int mi = 0; mi < 4; ++mi)
#pragma unroll
      for (int ni = 0; ni < 4; ++ni)
        acc[mi][ni] = __builtin_amdgcn_wmma_f32_16x16x32_bf16(
            false, af[mi].v, false, bfr[ni].v, (short)0, acc[mi][ni], false, false);
  }

#pragma unroll
  for (int mi = 0; mi < 4; ++mi) {
#pragma unroll
    for (int ni = 0; ni < 4; ++ni) {
      const int col = tn * 64 + ni * 16 + lr;
      if (col < N) {
        const float bv = bias ? bias[col] : 0.0f;
#pragma unroll
        for (int r = 0; r < 8; ++r) {
          const int row = tm * 64 + mi * 16 + r + hi * 8;
          if (row < M) C[(size_t)row * N + col] = acc[mi][ni][r] + bv;
        }
      }
    }
  }
}

// ---------------- weight convert + transpose: Wt[n*K+k] = bf16(W[k*N+n]) ----
__global__ void convT_bf16(const float* __restrict__ W, bf16* __restrict__ Wt,
                           int K, int N) {
  size_t i = (size_t)blockIdx.x * blockDim.x + threadIdx.x;
  if (i >= (size_t)K * N) return;
  size_t n = i / K, k = i % K;
  Wt[i] = (bf16)W[k * (size_t)N + n];
}

// ---------------- RMSNorm (optionally emits f32 and/or bf16) ----------------
__global__ __launch_bounds__(256)
void rmsnorm_kernel(const float* __restrict__ x, const float* __restrict__ w,
                    float* __restrict__ outf, bf16* __restrict__ outb, int ncols) {
  const int row = blockIdx.x;
  const float* xr = x + (size_t)row * ncols;
  __shared__ float red[256];
  float s = 0.f;
  for (int c = threadIdx.x; c < ncols; c += blockDim.x) { float v = xr[c]; s += v * v; }
  red[threadIdx.x] = s; __syncthreads();
  for (int off = 128; off > 0; off >>= 1) {
    if (threadIdx.x < off) red[threadIdx.x] += red[threadIdx.x + off];
    __syncthreads();
  }
  const float rs = rsqrtf(red[0] / ncols + EPS);
  for (int c = threadIdx.x; c < ncols; c += blockDim.x) {
    float v = xr[c] * rs * w[c];
    if (outf) outf[(size_t)row * ncols + c] = v;
    if (outb) outb[(size_t)row * ncols + c] = (bf16)v;
  }
}

// ---------------- depthwise causal conv (K=4) + SiLU ------------------------
__global__ void conv_silu_kernel(const float* __restrict__ proj,
                                 const float* __restrict__ cw,
                                 const float* __restrict__ cb,
                                 float* __restrict__ xbc) {
  size_t i = (size_t)blockIdx.x * blockDim.x + threadIdx.x;
  if (i >= (size_t)Lm * CONV_DIM) return;
  int t = (int)(i / CONV_DIM), c = (int)(i % CONV_DIM);
  float acc = cb[c];
#pragma unroll
  for (int kk = 0; kk < KSIZE; ++kk) {
    int ts = t - (KSIZE - 1) + kk;
    if (ts >= 0) acc += proj[(size_t)ts * D_INPROJ + MINT + c] * cw[c * KSIZE + kk];
  }
  xbc[i] = acc / (1.f + __expf(-acc));
}

// ---------------- dt = softplus(raw + bias) ---------------------------------
__global__ void dt_kernel(const float* __restrict__ proj,
                          const float* __restrict__ dtb, float* __restrict__ dt) {
  int i = blockIdx.x * blockDim.x + threadIdx.x;
  if (i >= Lm * NH) return;
  int t = i / NH, h = i % NH;
  float x = proj[(size_t)t * D_INPROJ + MINT + CONV_DIM + h] + dtb[h];
  dt[i] = (x > 20.f) ? x : log1pf(__expf(x));
}

// ---------------- sequential SSM scan: 1 block per head ---------------------
__global__ __launch_bounds__(256)
void scan_kernel(const float* __restrict__ xbc, const float* __restrict__ dt,
                 const float* __restrict__ alog, float* __restrict__ ys) {
  const int h = blockIdx.x, tid = threadIdx.x;
  const int d = tid >> 2, n0 = (tid & 3) << 6, g = h >> 2;
  const float Aneg = -__expf(alog[h]);
  __shared__ float xt[64], bt[256], ct[256], red[256];
  float hst[64];
#pragma unroll
  for (int i = 0; i < 64; ++i) hst[i] = 0.f;
  for (int t = 0; t < Lm; ++t) {
    const float* row = xbc + (size_t)t * CONV_DIM;
    if (tid < 64) xt[tid] = row[h * 64 + tid];
    bt[tid] = row[MINT + g * STATE + tid];
    ct[tid] = row[MINT + G * STATE + g * STATE + tid];
    __syncthreads();
    const float dtt = dt[t * NH + h];
    const float dA = __expf(dtt * Aneg);
    const float coef = dtt * xt[d];
    float acc = 0.f;
#pragma unroll
    for (int i = 0; i < 64; ++i) {
      float hv = hst[i] * dA + coef * bt[n0 + i];
      hst[i] = hv; acc += hv * ct[n0 + i];
    }
    red[tid] = acc; __syncthreads();
    if ((tid & 3) == 0)
      ys[(size_t)t * H + h * 64 + d] = red[tid] + red[tid + 1] + red[tid + 2] + red[tid + 3];
    __syncthreads();
  }
}

// ---------------- y = (ys + D*xs)*silu(z); gated RMSNorm -> bf16 -------------
__global__ __launch_bounds__(256)
void gate_norm_kernel(const float* __restrict__ ys, const float* __restrict__ xbc,
                      const float* __restrict__ proj, const float* __restrict__ Dp,
                      const float* __restrict__ mnw, bf16* __restrict__ ybf) {
  const int t = blockIdx.x;
  __shared__ float red[256];
  float vals[4], s = 0.f;
#pragma unroll
  for (int ii = 0; ii < 4; ++ii) {
    int c = threadIdx.x + ii * 256;
    float xsv = xbc[(size_t)t * CONV_DIM + c];
    float z = proj[(size_t)t * D_INPROJ + c];
    float v = (ys[(size_t)t * H + c] + Dp[c >> 6] * xsv) * (z / (1.f + __expf(-z)));
    vals[ii] = v; s += v * v;
  }
  red[threadIdx.x] = s; __syncthreads();
  for (int off = 128; off > 0; off >>= 1) {
    if (threadIdx.x < off) red[threadIdx.x] += red[threadIdx.x + off];
    __syncthreads();
  }
  const float rs = rsqrtf(red[0] / H + EPS);
#pragma unroll
  for (int ii = 0; ii < 4; ++ii) {
    int c = threadIdx.x + ii * 256;
    ybf[(size_t)t * H + c] = (bf16)(vals[ii] * rs * mnw[c]);
  }
}

// ---------------- hc = [hn[:128]; mem_out[-1]; hn[1792:]] -> bf16 ------------
__global__ void build_hc(const float* __restrict__ hn, const float* __restrict__ mo,
                         bf16* __restrict__ hcb) {
  int i = blockIdx.x * blockDim.x + threadIdx.x;
  if (i >= Tc * H) return;
  int t = i / H, c = i % H;
  float v;
  if (t < SINK)      v = hn[(size_t)t * H + c];
  else if (t == SINK) v = mo[(size_t)(Lm - 1) * H + c];
  else               v = hn[(size_t)(t + 1663) * H + c];
  hcb[i] = (bf16)v;
}

__global__ void build_cs(const float* __restrict__ cs, const float* __restrict__ sn,
                         float* __restrict__ cc, float* __restrict__ sc) {
  int i = blockIdx.x * blockDim.x + threadIdx.x;
  if (i >= Tc * HD) return;
  int t = i / HD, d = i % HD;
  int src = (t <= SINK) ? t : t + 1663;
  cc[i] = cs[src * HD + d];
  sc[i] = sn[src * HD + d];
}

// ---------------- RoPE in place; one thread per (t,h,pair) ------------------
__global__ void rope_kernel(float* __restrict__ qk, const float* __restrict__ cc,
                            const float* __restrict__ sc, int nh) {
  int i = blockIdx.x * blockDim.x + threadIdx.x;
  if (i >= Tc * nh * 32) return;
  int p = i % 32, rest = i / 32;
  int t = rest / nh;
  float* vp = qk + (size_t)rest * 64;
  float a = vp[p], b = vp[p + 32];
  float c1 = cc[t * 64 + p],      s1 = sc[t * 64 + p];
  float c2 = cc[t * 64 + p + 32], s2 = sc[t * 64 + p + 32];
  vp[p]      = a * c1 - b * s1;
  vp[p + 32] = b * c2 + a * s2;
}

// ---------------- causal attention: block per (query, head) -----------------
__global__ __launch_bounds__(128)
void attn_kernel(const float* __restrict__ q, const float* __restrict__ k,
                 const float* __restrict__ v, float* __restrict__ ao) {
  const int qi = blockIdx.x, h = blockIdx.y, tid = threadIdx.x;
  const int kv = h >> 2;
  __shared__ float sc[Tc], red[128], qs[64];
  if (tid < 64) qs[tid] = q[(size_t)qi * (NH * HD) + h * HD + tid];
  __syncthreads();
  const int nk = qi + 1;
  float mx = -1e30f;
  for (int j = tid; j < nk; j += 128) {
    const float* kr = k + (size_t)j * (NKV * HD) + kv * HD;
    float s = 0.f;
    for (int d = 0; d < 64; ++d) s += qs[d] * kr[d];
    s *= 0.125f;
    sc[j] = s; mx = fmaxf(mx, s);
  }
  red[tid] = mx; __syncthreads();
  for (int off = 64; off > 0; off >>= 1) {
    if (tid < off) red[tid] = fmaxf(red[tid], red[tid + off]);
    __syncthreads();
  }
  mx = red[0]; __syncthreads();
  float se = 0.f;
  for (int j = tid; j < nk; j += 128) { float e = __expf(sc[j] - mx); sc[j] = e; se += e; }
  red[tid] = se; __syncthreads();
  for (int off = 64; off > 0; off >>= 1) {
    if (tid < off) red[tid] += red[tid + off];
    __syncthreads();
  }
  const float inv = 1.f / red[0];
  __syncthreads();
  if (tid < 64) {
    float o = 0.f;
    for (int j = 0; j < nk; ++j) o += sc[j] * v[(size_t)j * (NKV * HD) + kv * HD + tid];
    ao[(size_t)qi * H + h * HD + tid] = o * inv;
  }
}

// ---------------- ao_full = [ao[:128]; mem_out; ao[129:]] -> bf16 ------------
__global__ void build_aofull(const float* __restrict__ ao, const float* __restrict__ mo,
                             bf16* __restrict__ out) {
  size_t i = (size_t)blockIdx.x * blockDim.x + threadIdx.x;
  if (i >= (size_t)T * H) return;
  int t = (int)(i / H), c = (int)(i % H);
  float v;
  if (t < SINK)           v = ao[(size_t)t * H + c];
  else if (t < T - WIN)   v = mo[(size_t)(t - SINK) * H + c];
  else                    v = ao[(size_t)(t - 1663) * H + c];
  out[i] = (bf16)v;
}

__global__ void add_kernel(const float* __restrict__ a, const float* __restrict__ b,
                           float* __restrict__ c, size_t n) {
  size_t i = (size_t)blockIdx.x * blockDim.x + threadIdx.x;
  if (i < n) c[i] = a[i] + b[i];
}

__global__ void silu_mul_kernel(const float* __restrict__ g, const float* __restrict__ u,
                                bf16* __restrict__ out, size_t n) {
  size_t i = (size_t)blockIdx.x * blockDim.x + threadIdx.x;
  if (i >= n) return;
  float gv = g[i];
  out[i] = (bf16)((gv / (1.f + __expf(-gv))) * u[i]);
}

// ============================ launch ========================================
extern "C" void kernel_launch(void* const* d_in, const int* in_sizes, int n_in,
                              void* d_out, int out_size, void* d_ws, size_t ws_size,
                              hipStream_t stream) {
  const float* x     = (const float*)d_in[0];
  const float* cosA  = (const float*)d_in[1];
  const float* sinA  = (const float*)d_in[2];
  const float* ln1   = (const float*)d_in[3];
  const float* qw    = (const float*)d_in[4];
  const float* qb    = (const float*)d_in[5];
  const float* kw    = (const float*)d_in[6];
  const float* kb    = (const float*)d_in[7];
  const float* vw    = (const float*)d_in[8];
  const float* vb    = (const float*)d_in[9];
  const float* ow    = (const float*)d_in[10];
  const float* ipw   = (const float*)d_in[11];
  const float* cw    = (const float*)d_in[12];
  const float* cb    = (const float*)d_in[13];
  const float* dtb   = (const float*)d_in[14];
  const float* alog  = (const float*)d_in[15];
  const float* Dp    = (const float*)d_in[16];
  const float* mnw   = (const float*)d_in[17];
  const float* moutw = (const float*)d_in[18];
  const float* ln2   = (const float*)d_in[19];
  const float* gatew = (const float*)d_in[20];
  const float* upw   = (const float*)d_in[21];
  const float* downw = (const float*)d_in[22];
  float* out = (float*)d_out;

  size_t off = 0;
  auto alloc = [&](size_t bytes) -> char* {
    char* p = (char*)d_ws + off;
    off = (off + bytes + 255) & ~(size_t)255;
    return p;
  };
  float* hn      = (float*)alloc((size_t)T * H * 4);
  bf16*  hnb     = (bf16*) alloc((size_t)T * H * 2);
  bf16*  ipT     = (bf16*) alloc((size_t)D_INPROJ * H * 2);
  bf16*  moutT   = (bf16*) alloc((size_t)H * H * 2);
  bf16*  qT      = (bf16*) alloc((size_t)H * H * 2);
  bf16*  kT      = (bf16*) alloc((size_t)(NKV * HD) * H * 2);
  bf16*  vT      = (bf16*) alloc((size_t)(NKV * HD) * H * 2);
  bf16*  oT      = (bf16*) alloc((size_t)H * H * 2);
  bf16*  gateT   = (bf16*) alloc((size_t)INTER * H * 2);
  bf16*  upT     = (bf16*) alloc((size_t)INTER * H * 2);
  bf16*  downT   = (bf16*) alloc((size_t)H * INTER * 2);
  float* proj    = (float*)alloc((size_t)Lm * D_INPROJ * 4);
  float* xbc     = (float*)alloc((size_t)Lm * CONV_DIM * 4);
  float* dtbuf   = (float*)alloc((size_t)Lm * NH * 4);
  float* ys      = (float*)alloc((size_t)Lm * H * 4);
  bf16*  ybf     = (bf16*) alloc((size_t)Lm * H * 2);
  float* memout  = (float*)alloc((size_t)Lm * H * 4);
  bf16*  hcb     = (bf16*) alloc((size_t)Tc * H * 2);
  float* cosc    = (float*)alloc((size_t)Tc * HD * 4);
  float* sinc    = (float*)alloc((size_t)Tc * HD * 4);
  float* qbuf    = (float*)alloc((size_t)Tc * NH * HD * 4);
  float* kbuf    = (float*)alloc((size_t)Tc * NKV * HD * 4);
  float* vbuf    = (float*)alloc((size_t)Tc * NKV * HD * 4);
  float* ao      = (float*)alloc((size_t)Tc * H * 4);
  bf16*  aofb    = (bf16*) alloc((size_t)T * H * 2);
  float* aproj   = (float*)alloc((size_t)T * H * 4);
  float* h1      = (float*)alloc((size_t)T * H * 4);
  bf16*  mlpinb  = (bf16*) alloc((size_t)T * H * 2);
  float* gbuf    = (float*)alloc((size_t)T * INTER * 4);
  float* ubuf    = (float*)alloc((size_t)T * INTER * 4);
  bf16*  actb    = (bf16*) alloc((size_t)T * INTER * 2);
  float* mlpout  = (float*)alloc((size_t)T * H * 4);
  (void)ws_size; (void)in_sizes; (void)n_in; (void)out_size;

  auto gemm = [&](const bf16* A, const bf16* Bt, const float* bias, float* C,
                  int M, int N, int K) {
    int tm = (M + 63) / 64, tn = (N + 63) / 64;
    int tiles = tm * tn;
    gemm_bf16_wmma<<<dim3((tiles + 7) / 8), dim3(256), 0, stream>>>(
        A, Bt, bias, C, M, N, K, tm, tn);
  };
  auto convT = [&](const float* W, bf16* Wt, int K, int N) {
    size_t tot = (size_t)K * N;
    convT_bf16<<<dim3((unsigned)((tot + 255) / 256)), dim3(256), 0, stream>>>(W, Wt, K, N);
  };

  // 1) pre-norm
  rmsnorm_kernel<<<dim3(T), dim3(256), 0, stream>>>(x, ln1, hn, hnb, H);
  // 2) weights -> bf16 transposed
  convT(ipw,   ipT,   H, D_INPROJ);
  convT(moutw, moutT, H, H);
  convT(qw,    qT,    H, H);
  convT(kw,    kT,    H, NKV * HD);
  convT(vw,    vT,    H, NKV * HD);
  convT(ow,    oT,    H, H);
  convT(gatew, gateT, H, INTER);
  convT(upw,   upT,   H, INTER);
  convT(downw, downT, INTER, H);
  // 3) mamba: in-proj GEMM over evicted rows
  gemm(hnb + (size_t)SINK * H, ipT, nullptr, proj, Lm, D_INPROJ, H);
  // 4) conv + silu ; dt
  {
    size_t tot = (size_t)Lm * CONV_DIM;
    conv_silu_kernel<<<dim3((unsigned)((tot + 255) / 256)), dim3(256), 0, stream>>>(proj, cw, cb, xbc);
  }
  dt_kernel<<<dim3((Lm * NH + 255) / 256), dim3(256), 0, stream>>>(proj, dtb, dtbuf);
  // 5) sequential SSM scan
  scan_kernel<<<dim3(NH), dim3(256), 0, stream>>>(xbc, dtbuf, alog, ys);
  // 6) gate + mamba RMSNorm -> bf16 ; out-proj
  gate_norm_kernel<<<dim3(Lm), dim3(256), 0, stream>>>(ys, xbc, proj, Dp, mnw, ybf);
  gemm(ybf, moutT, nullptr, memout, Lm, H, H);
  // 7) compressed sequence + rope tables
  build_hc<<<dim3((Tc * H + 255) / 256), dim3(256), 0, stream>>>(hn, memout, hcb);
  build_cs<<<dim3((Tc * HD + 255) / 256), dim3(256), 0, stream>>>(cosA, sinA, cosc, sinc);
  // 8) qkv projections
  gemm(hcb, qT, qb, qbuf, Tc, NH * HD, H);
  gemm(hcb, kT, kb, kbuf, Tc, NKV * HD, H);
  gemm(hcb, vT, vb, vbuf, Tc, NKV * HD, H);
  // 9) rope
  rope_kernel<<<dim3((Tc * NH * 32 + 255) / 256), dim3(256), 0, stream>>>(qbuf, cosc, sinc, NH);
  rope_kernel<<<dim3((Tc * NKV * 32 + 255) / 256), dim3(256), 0, stream>>>(kbuf, cosc, sinc, NKV);
  // 10) attention
  attn_kernel<<<dim3(Tc, NH), dim3(128), 0, stream>>>(qbuf, kbuf, vbuf, ao);
  // 11) splice mem_out back, o-proj, residual
  {
    size_t tot = (size_t)T * H;
    build_aofull<<<dim3((unsigned)((tot + 255) / 256)), dim3(256), 0, stream>>>(ao, memout, aofb);
  }
  gemm(aofb, oT, nullptr, aproj, T, H, H);
  add_kernel<<<dim3((unsigned)(((size_t)T * H + 255) / 256)), dim3(256), 0, stream>>>(x, aproj, h1, (size_t)T * H);
  // 12) MLP
  rmsnorm_kernel<<<dim3(T), dim3(256), 0, stream>>>(h1, ln2, nullptr, mlpinb, H);
  gemm(mlpinb, gateT, nullptr, gbuf, T, INTER, H);
  gemm(mlpinb, upT,   nullptr, ubuf, T, INTER, H);
  {
    size_t tot = (size_t)T * INTER;
    silu_mul_kernel<<<dim3((unsigned)((tot + 255) / 256)), dim3(256), 0, stream>>>(gbuf, ubuf, actb, tot);
  }
  gemm(actb, downT, nullptr, mlpout, T, H, INTER);
  add_kernel<<<dim3((unsigned)(((size_t)T * H + 255) / 256)), dim3(256), 0, stream>>>(h1, mlpout, out, (size_t)T * H);
}
